// Siamese_68616397521413
// MI455X (gfx1250) — matrix operand order repaired and checked by reading, not compiled, not run on hardware
//
#include <hip/hip_runtime.h>
#include <hip/hip_bf16.h>

typedef __bf16 bf16_t;
typedef __attribute__((ext_vector_type(16))) __bf16 v16bf;
typedef __attribute__((ext_vector_type(8)))  __bf16 v8bf;
typedef __attribute__((ext_vector_type(8)))  float  v8f;
typedef __attribute__((ext_vector_type(4)))  int    v4i;
typedef __attribute__((ext_vector_type(8)))  int    v8i;
typedef __attribute__((ext_vector_type(4)))  unsigned int v4u;

#if __has_builtin(__builtin_amdgcn_tensor_load_to_lds) && \
    __has_builtin(__builtin_amdgcn_s_wait_tensorcnt)
#define USE_TDM 1
#endif

// ---------------------------------------------------------------- utilities
__global__ void k_zero(float* __restrict__ p, int n) {
    int i = blockIdx.x * blockDim.x + threadIdx.x;
    if (i < n) p[i] = 0.f;
}

__global__ void k_tobf16(const float* __restrict__ in, bf16_t* __restrict__ out,
                         long long n) {
    long long i = blockIdx.x * (long long)blockDim.x + threadIdx.x;
    if (i < n) out[i] = (bf16_t)in[i];
}

// ----------------------------------------------- BN batch statistics (sum,sumsq)
__global__ void k_stats(const float* __restrict__ x, float* __restrict__ sums,
                        int B, int C, int HW) {
    int c = blockIdx.x;
    __shared__ float s1[256];
    __shared__ float s2[256];
    float a = 0.f, q = 0.f;
    for (int n = 0; n < B; ++n) {
        const float* p = x + (long long)(n * C + c) * HW;
        for (int i = blockIdx.y * blockDim.x + threadIdx.x; i < HW;
             i += gridDim.y * blockDim.x) {
            float v = p[i];
            a += v; q += v * v;
        }
    }
    s1[threadIdx.x] = a; s2[threadIdx.x] = q;
    __syncthreads();
    for (int off = 128; off > 0; off >>= 1) {
        if ((int)threadIdx.x < off) {
            s1[threadIdx.x] += s1[threadIdx.x + off];
            s2[threadIdx.x] += s2[threadIdx.x + off];
        }
        __syncthreads();
    }
    if (threadIdx.x == 0) {
        atomicAdd(&sums[c],     s1[0]);
        atomicAdd(&sums[C + c], s2[0]);
    }
}

// -------------------------------- fused BN(+ReLU)(+MaxPool3x3s2) -> bf16 output
__global__ void k_bn_act(const float* __restrict__ x, const float* __restrict__ sums,
                         const float* __restrict__ g, const float* __restrict__ bb,
                         bf16_t* __restrict__ out,
                         int B, int C, int H, int W, int Ho, int Wo,
                         int do_relu, int do_pool, float invcnt) {
    long long idx = blockIdx.x * (long long)blockDim.x + threadIdx.x;
    long long total = (long long)B * C * Ho * Wo;
    if (idx >= total) return;
    int ow = (int)(idx % Wo);
    long long t = idx / Wo;
    int oh = (int)(t % Ho); t /= Ho;
    int c  = (int)(t % C);
    int b  = (int)(t / C);
    float mean = sums[c] * invcnt;
    float var  = sums[C + c] * invcnt - mean * mean;
    float rs   = rsqrtf(var + 1e-5f);
    float gg   = g[c] * rs;
    float bv   = bb[c] - mean * gg;
    const float* p = x + (long long)(b * C + c) * H * W;
    int win = do_pool ? 3 : 1, st = do_pool ? 2 : 1;
    float m = -3.4e38f;
    for (int u = 0; u < win; ++u)
        for (int v = 0; v < win; ++v) {
            float val = p[(oh * st + u) * W + (ow * st + v)] * gg + bv;
            if (do_relu) val = fmaxf(val, 0.f);
            m = fmaxf(m, val);
        }
    out[idx] = (bf16_t)m;
}

// --------------------- pack OIHW fp32 weights into WMMA A-fragment bf16 layout
__global__ void k_pack(const float* __restrict__ w, bf16_t* __restrict__ pw,
                       int Cout, int Kreal, int Kpad) {
    int Mtiles = Cout >> 4;
    long long t = blockIdx.x * (long long)blockDim.x + threadIdx.x;
    long long total = (long long)Cout * Kpad;
    if (t >= total) return;
    int e    = (int)(t & 15);
    int lane = (int)((t >> 4) & 31);
    long long q = t >> 9;
    int mtile = (int)(q % Mtiles);
    int kc    = (int)(q / Mtiles);
    int m   = mtile * 16 + (lane & 15);
    int grp = ((e >> 3) << 1) | (lane >> 4);
    int k   = kc * 32 + grp * 8 + (e & 7);
    pw[t] = (k < Kreal) ? (bf16_t)w[(long long)m * Kreal + k] : (bf16_t)0.f;
}

// -------------------- implicit-GEMM conv via v_wmma_f32_16x16x32_bf16 ---------
// 4 waves / block share a double-buffered 32x16 staged B (im2col) tile; each
// wave owns a PAIR of M-tiles (all layer Mtiles are even) -> 8 WMMAs amortize
// one B staging.  Square inputs (W==H, Wo==Ho).
template <int KH, int KW, int S>
__global__ void __launch_bounds__(128)
k_conv_wmma(const bf16_t* __restrict__ act, const bf16_t* __restrict__ pw,
            const float* __restrict__ bias, float* __restrict__ out,
            int B, int Cin, int H, int Cout, int Ho, int Kreal) {
    const int tid    = threadIdx.x;
    const int lane   = tid & 31;
    const int wv     = tid >> 5;
    const int ntile  = blockIdx.x;
    const int Mtiles = Cout >> 4;
    const int KC     = (Kreal + 31) >> 5;
    const int Ptot   = B * Ho * Ho;
    const int HW     = H * H;

    __shared__ __align__(16) bf16_t Btr[2][16][40];  // double-buffered [n][K']
    __shared__ int pbase[16];

    if (tid < 16) {
        int p = ntile * 16 + tid;
        if (p >= Ptot) p = Ptot - 1;
        int ow = p % Ho;
        int t2 = p / Ho;
        int oh = t2 % Ho;
        int b  = t2 / Ho;
        pbase[tid] = (b * Cin) * HW + (oh * S) * H + ow * S;
    }
    __syncthreads();

    int  mtB = (blockIdx.y * 4 + wv) * 2;    // this wave's M-tile pair
    bool mok = (mtB < Mtiles);
    int  mtc = mok ? mtB : 0;
    const long long astride = (long long)Mtiles * 512;
    const bf16_t* abase = pw + ((long long)mtc * 32 + lane) * 16;

    const int Kp = tid >> 2;        // this thread's B row K' in 0..31
    const int j0 = (tid & 3) * 4;   // this thread's 4 pixels
    const int n  = lane & 15;
    const int k0 = (lane >> 4) << 3;

    v8f acc0 = {}, acc1 = {};
    for (int kc = 0; kc < KC; ++kc) {
        const int buf = kc & 1;
        // ---- cooperative B-tile staging (independent gathers, then stores)
        int k = kc * 32 + Kp;
        bf16_t t0, t1, t2v, t3;
        if (k < Kreal) {
            int ci = k / (KH * KW);
            int r  = k % (KH * KW);
            int rowoff = ci * HW + (r / KW) * H + (r % KW);
            t0  = act[pbase[j0 + 0] + rowoff];
            t1  = act[pbase[j0 + 1] + rowoff];
            t2v = act[pbase[j0 + 2] + rowoff];
            t3  = act[pbase[j0 + 3] + rowoff];
        } else {
            t0 = t1 = t2v = t3 = (bf16_t)0.f;
        }
        Btr[buf][j0 + 0][Kp] = t0;
        Btr[buf][j0 + 1][Kp] = t1;
        Btr[buf][j0 + 2][Kp] = t2v;
        Btr[buf][j0 + 3][Kp] = t3;

        // ---- A fragments for the M-tile pair (adjacent 32B loads)
        const bf16_t* ap = abase + (long long)kc * astride;
        v16bf a0 = *reinterpret_cast<const v16bf*>(ap);
        v16bf a1 = *reinterpret_cast<const v16bf*>(ap + 512);
        if (kc + 1 < KC)
            __builtin_prefetch((const void*)(ap + astride), 0, 3);
        __syncthreads();

        // ---- B fragment: two 16B LDS vector loads per lane
        v8bf blo = *reinterpret_cast<const v8bf*>(&Btr[buf][n][k0]);
        v8bf bhi = *reinterpret_cast<const v8bf*>(&Btr[buf][n][k0 + 16]);
        v16bf bfrag = __builtin_shufflevector(blo, bhi, 0, 1, 2, 3, 4, 5, 6, 7,
                                              8, 9, 10, 11, 12, 13, 14, 15);
        acc0 = __builtin_amdgcn_wmma_f32_16x16x32_bf16(
                   false, a0, false, bfrag, (short)0, acc0, false, false);
        acc1 = __builtin_amdgcn_wmma_f32_16x16x32_bf16(
                   false, a1, false, bfrag, (short)0, acc1, false, false);
    }

    // ---- epilogue: D VGPR r -> M = r + 8*(lane>=16), N = lane&15
    int p = ntile * 16 + n;
    if (mok && p < Ptot) {
        int ow = p % Ho;
        int t2 = p / Ho;
        int oh = t2 % Ho;
        int b  = t2 / Ho;
        long long base = ((long long)(b * Cout) * Ho + oh) * Ho + ow;
        long long cstr = (long long)Ho * Ho;
        #pragma unroll
        for (int r8 = 0; r8 < 8; ++r8) {
            int m   = r8 + ((lane >> 4) << 3);
            int co0 = mtB * 16 + m;
            int co1 = co0 + 16;
            out[base + (long long)co0 * cstr] = acc0[r8] + bias[co0];
            out[base + (long long)co1 * cstr] = acc1[r8] + bias[co1];
        }
    }
}

// ------------- pack o1 into WMMA A-fragment order for the correlation --------
// Layout: [mtile(19)][kc(4608)][lane(32)][e(16)].  Fully parallel gathers.
__global__ void k_packA(const bf16_t* __restrict__ o1, bf16_t* __restrict__ pa,
                        long long total) {
    long long t = blockIdx.x * (long long)blockDim.x + threadIdx.x;
    if (t >= total) return;
    int e    = (int)(t & 15);
    int lane = (int)((t >> 4) & 31);
    long long q = t >> 9;
    int kc    = (int)(q % 4608);
    int mtile = (int)(q / 4608);
    int m = lane & 15;
    int p = mtile * 16 + m;
    if (p > 288) p = 288;
    int oh = p / 17, ow = p % 17;
    int grp = ((e >> 3) << 1) | (lane >> 4);
    int k   = kc * 32 + grp * 8 + (e & 7);
    int v6  = k % 6;
    int t1  = k / 6;
    int u   = t1 % 6;
    int bc  = t1 / 6;                    // b*128 + c
    pa[t] = o1[((long long)bc * 22 + (oh + u)) * 22 + (ow + v6)];
}

// ---------------- cross-correlation via WMMA, K split over 64 blocks ----------
// M = 289 output pixels (19 tiles), K = 147456 (4608 chunks); A pre-packed,
// o2 slice staged into LDS via the Tensor Data Mover.
__global__ void __launch_bounds__(32)
k_corr(const bf16_t* __restrict__ pa, const bf16_t* __restrict__ o2,
       float* __restrict__ corr) {
    const int lane   = threadIdx.x;
    const int mtile  = blockIdx.x;      // 19
    const int ks     = blockIdx.y;      // 64 K-splits
    const int CHUNKS = 72;              // 4608 / 64
    const int kbase  = ks * CHUNKS * 32;

    __shared__ __align__(16) bf16_t o2s[CHUNKS * 32];

#if defined(USE_TDM)
    {   // Tensor Data Mover: 1-D tile (4608B) global -> LDS
        unsigned lds_off = (unsigned)(unsigned long long)(const void*)&o2s[0];
        unsigned long long ga = (unsigned long long)(const void*)(o2 + kbase);
        v4u g0;
        g0[0] = 1u;                                   // count=1, user mode
        g0[1] = lds_off;                              // lds_addr
        g0[2] = (unsigned)(ga & 0xffffffffu);
        g0[3] = (unsigned)(((ga >> 32) & 0x1ffffffu) | (2u << 30)); // type=2
        v8i g1;
        g1[0] = (1 << 16);                            // data_size=1 (2 bytes)
        g1[1] = (int)(2304u << 16);                   // tensor_dim0[15:0]
        g1[2] = (1 << 16);                            // tensor_dim1 = 1
        g1[3] = (int)(2304u << 16);                   // tile_dim0 = 2304
        g1[4] = 1;                                    // tile_dim1 = 1
        g1[5] = 2304;                                 // tensor_dim0_stride
        g1[6] = 0;
        g1[7] = 0;
        v4i zz4 = {0, 0, 0, 0};
        v8i zz8 = {0, 0, 0, 0, 0, 0, 0, 0};
        __builtin_amdgcn_tensor_load_to_lds(g0, g1, zz4, zz4, zz8, 0);
        __builtin_amdgcn_s_wait_tensorcnt(0);
    }
#else
    for (int i = lane; i < CHUNKS * 32; i += 32) o2s[i] = o2[kbase + i];
#endif
    __syncthreads();

    v8f acc = {};
    const int k0 = (lane >> 4) << 3;
    const bf16_t* ap = pa + ((long long)mtile * 4608 + ks * CHUNKS) * 512
                          + (long long)lane * 16;
    for (int kc = 0; kc < CHUNKS; ++kc) {
        v16bf afrag = *reinterpret_cast<const v16bf*>(ap);
        ap += 512;
        if (kc + 1 < CHUNKS)
            __builtin_prefetch((const void*)ap, 0, 3);
        v8bf blo = *reinterpret_cast<const v8bf*>(&o2s[kc * 32 + k0]);
        v8bf bhi = *reinterpret_cast<const v8bf*>(&o2s[kc * 32 + k0 + 16]);
        v16bf bfrag = __builtin_shufflevector(blo, bhi, 0, 1, 2, 3, 4, 5, 6, 7,
                                              8, 9, 10, 11, 12, 13, 14, 15);
        acc = __builtin_amdgcn_wmma_f32_16x16x32_bf16(
                  false, afrag, false, bfrag, (short)0, acc, false, false);
    }
    if ((lane & 15) == 0) {
        const float inv = 1.0f / 147456.0f;
        #pragma unroll
        for (int r8 = 0; r8 < 8; ++r8) {
            int mm = r8 + ((lane >> 4) << 3);
            int pp = mtile * 16 + mm;
            if (pp < 289) atomicAdd(&corr[pp], acc[r8] * inv);
        }
    }
}

__global__ void k_bcast(const float* __restrict__ corr, float* __restrict__ out, int n) {
    int i = blockIdx.x * blockDim.x + threadIdx.x;
    if (i < n) out[i] = corr[i % 289];
}

// ---------------------------------------------------------------- host driver
static inline int cdiv(long long a, int b) { return (int)((a + b - 1) / b); }

extern "C" void kernel_launch(void* const* d_in, const int* in_sizes, int n_in,
                              void* d_out, int out_size, void* d_ws, size_t ws_size,
                              hipStream_t stream) {
    (void)in_sizes; (void)n_in; (void)out_size; (void)ws_size;
    const float* input1 = (const float*)d_in[0];
    const float* input2 = (const float*)d_in[1];
    const float* conv_w[5] = {(const float*)d_in[2], (const float*)d_in[4],
                              (const float*)d_in[6], (const float*)d_in[8],
                              (const float*)d_in[10]};
    const float* conv_b[5] = {(const float*)d_in[3], (const float*)d_in[5],
                              (const float*)d_in[7], (const float*)d_in[9],
                              (const float*)d_in[11]};
    const float* bn_g[5] = {(const float*)d_in[12], (const float*)d_in[14],
                            (const float*)d_in[16], (const float*)d_in[18],
                            (const float*)d_in[20]};
    const float* bn_b[5] = {(const float*)d_in[13], (const float*)d_in[15],
                            (const float*)d_in[17], (const float*)d_in[19],
                            (const float*)d_in[21]};

    char* ws = (char*)d_ws;
    size_t off = 0;
    auto carve = [&](size_t bytes) -> char* {
        char* p = ws + off;
        off = (off + bytes + 255) & ~(size_t)255;
        return p;
    };
    float*  convbuf = (float*)carve(sizeof(float) * (size_t)48 * 1024 * 1024);
    bf16_t* actA    = (bf16_t*)carve(sizeof(bf16_t) * (size_t)13 * 1024 * 1024);
    bf16_t* actB    = (bf16_t*)carve(sizeof(bf16_t) * (size_t)8 * 1024 * 1024);
    bf16_t* in1b    = (bf16_t*)carve(sizeof(bf16_t) * (size_t)32 * 3 * 255 * 255);
    bf16_t* in2b    = (bf16_t*)carve(sizeof(bf16_t) * (size_t)32 * 3 * 127 * 127);
    bf16_t* o1      = (bf16_t*)carve(sizeof(bf16_t) * (size_t)32 * 128 * 22 * 22);
    bf16_t* o2      = (bf16_t*)carve(sizeof(bf16_t) * (size_t)32 * 128 * 6 * 6);
    const int CoutA[5] = {96, 256, 192, 192, 128};
    const int KrA[5]   = {3 * 121, 96 * 25, 256 * 9, 192 * 9, 192 * 9};
    bf16_t* pw[5];
    int Kpad[5];
    for (int i = 0; i < 5; ++i) {
        Kpad[i] = ((KrA[i] + 31) / 32) * 32;
        pw[i] = (bf16_t*)carve(sizeof(bf16_t) * (size_t)CoutA[i] * Kpad[i]);
    }
    float* stats = (float*)carve(sizeof(float) * 1024);
    float* corr  = (float*)carve(sizeof(float) * 512);

    // input conversion + weight packing
    long long n1 = (long long)32 * 3 * 255 * 255;
    long long n2 = (long long)32 * 3 * 127 * 127;
    k_tobf16<<<cdiv(n1, 256), 256, 0, stream>>>(input1, in1b, n1);
    k_tobf16<<<cdiv(n2, 256), 256, 0, stream>>>(input2, in2b, n2);
    for (int i = 0; i < 5; ++i) {
        long long tot = (long long)CoutA[i] * Kpad[i];
        k_pack<<<cdiv(tot, 256), 256, 0, stream>>>(conv_w[i], pw[i], CoutA[i],
                                                   KrA[i], Kpad[i]);
    }

    for (int br = 0; br < 2; ++br) {
        const bf16_t* inp = br ? in2b : in1b;
        const int H0  = br ? 127 : 255;
        const int h1  = (H0 - 11) / 2 + 1;   // 123 / 59
        const int hp1 = (h1 - 3) / 2 + 1;    // 61  / 29
        const int h2  = hp1 - 4;             // 57  / 25
        const int hp2 = (h2 - 3) / 2 + 1;    // 28  / 12
        const int h3  = hp2 - 2;             // 26  / 10
        const int h4  = h3 - 2;              // 24  / 8
        const int h5  = h4 - 2;              // 22  / 6
        bf16_t* obuf = br ? o2 : o1;

        // ---- layer 1 (11x11 s2) via WMMA with zero-padded K
        {
            long long Ptot = (long long)32 * h1 * h1;
            dim3 grid((unsigned)cdiv(Ptot, 16), (unsigned)cdiv((96 >> 4) / 2, 4));
            k_conv_wmma<11, 11, 2><<<grid, 128, 0, stream>>>(
                inp, pw[0], conv_b[0], convbuf, 32, 3, H0, 96, h1, KrA[0]);
            k_zero<<<1, 256, 0, stream>>>(stats, 2 * 96);
            k_stats<<<dim3(96, 8), 256, 0, stream>>>(convbuf, stats, 32, 96, h1 * h1);
            long long an = (long long)32 * 96 * hp1 * hp1;
            k_bn_act<<<cdiv(an, 256), 256, 0, stream>>>(
                convbuf, stats, bn_g[0], bn_b[0], actA, 32, 96, h1, h1, hp1, hp1,
                1, 1, 1.0f / (32.0f * h1 * h1));
        }
        // ---- layers 2..5
        struct LayerCfg { const bf16_t* in; bf16_t* outb; int Cin, Hin, Cout,
                          Hout, Hact, relu, pool; };
        LayerCfg L[4] = {
            {actA, actB, 96,  hp1, 256, h2, hp2, 1, 1},
            {actB, actA, 256, hp2, 192, h3, h3,  1, 0},
            {actA, actB, 192, h3,  192, h4, h4,  1, 0},
            {actB, obuf, 192, h4,  128, h5, h5,  0, 0}};
        for (int i = 0; i < 4; ++i) {
            long long Ptot = (long long)32 * L[i].Hout * L[i].Hout;
            dim3 grid((unsigned)cdiv(Ptot, 16),
                      (unsigned)cdiv((L[i].Cout >> 4) / 2, 4));
            if (i == 0)
                k_conv_wmma<5, 5, 1><<<grid, 128, 0, stream>>>(
                    L[i].in, pw[1], conv_b[1], convbuf, 32, L[i].Cin, L[i].Hin,
                    L[i].Cout, L[i].Hout, KrA[1]);
            else
                k_conv_wmma<3, 3, 1><<<grid, 128, 0, stream>>>(
                    L[i].in, pw[i + 1], conv_b[i + 1], convbuf, 32, L[i].Cin,
                    L[i].Hin, L[i].Cout, L[i].Hout, KrA[i + 1]);
            k_zero<<<1, 256, 0, stream>>>(stats, 2 * L[i].Cout);
            k_stats<<<dim3(L[i].Cout, 8), 256, 0, stream>>>(
                convbuf, stats, 32, L[i].Cout, L[i].Hout * L[i].Hout);
            long long an = (long long)32 * L[i].Cout * L[i].Hact * L[i].Hact;
            k_bn_act<<<cdiv(an, 256), 256, 0, stream>>>(
                convbuf, stats, bn_g[i + 1], bn_b[i + 1], L[i].outb,
                32, L[i].Cout, L[i].Hout, L[i].Hout, L[i].Hact, L[i].Hact,
                L[i].relu, L[i].pool, 1.0f / (32.0f * L[i].Hout * L[i].Hout));
        }
    }

    // ---- cross-correlation: pack A (reusing convbuf), WMMA with TDM-staged B
    bf16_t* packA = (bf16_t*)convbuf;       // 19*4608*512 bf16 ~ 90MB, fits
    long long paTot = (long long)19 * 4608 * 512;
    k_packA<<<cdiv(paTot, 256), 256, 0, stream>>>(o1, packA, paTot);
    k_zero<<<2, 256, 0, stream>>>(corr, 289);
    k_corr<<<dim3(19, 64), 32, 0, stream>>>(packA, o2, corr);
    k_bcast<<<cdiv(32 * 289, 256), 256, 0, stream>>>(corr, (float*)d_out, 32 * 289);
}
